// GNNRoutingNet_60447369724616
// MI455X (gfx1250) — matrix operand order repaired and checked by reading, not compiled, not run on hardware
//
#include <hip/hip_runtime.h>
#include <hip/hip_bf16.h>

#define B_    64
#define A_    50
#define S_    20
#define F_    32
#define N_    3600
#define EDG_  14400
#define ED_   128
#define NOUT_ 16
#define NE_   (EDG_ + N_)   // 18000 edges incl. self loops
#define ROWS_ (A_ + S_)     // 70
#define MTOT_ (B_ * N_)     // 230400 rows in the big GEMM

typedef __attribute__((ext_vector_type(8)))  __bf16 v8bf;
typedef __attribute__((ext_vector_type(16))) __bf16 v16bf;
typedef __attribute__((ext_vector_type(8)))  float  v8f;
typedef __attribute__((ext_vector_type(4)))  float  v4f;

static __device__ __forceinline__ float atomAddF(float* p, float v) {
    return __hip_atomic_fetch_add(p, v, __ATOMIC_RELAXED, __HIP_MEMORY_SCOPE_AGENT);
}

// low 32 bits of a generic pointer to __shared__ data == LDS byte offset
// (generic shared address = {SHARED_BASE, offset[31:0]}, ISA 10.2)
static __device__ __forceinline__ unsigned lds_off(const void* p) {
    return (unsigned)(size_t)p;
}

// per-lane async copy of 16 bytes global -> LDS, tracked by ASYNCcnt
static __device__ __forceinline__ void async_g2l_b128(unsigned lds, const float* g) {
    asm volatile("global_load_async_to_lds_b128 %0, %1, off"
                 :: "v"(lds), "v"((unsigned long long)(size_t)g)
                 : "memory");
}

// ---------------------------------------------------------------- utilities
__global__ void k_fill(float* __restrict__ p, size_t n, float v) {
    size_t i = (size_t)blockIdx.x * blockDim.x + threadIdx.x;
    size_t stride = (size_t)gridDim.x * blockDim.x;
    for (; i < n; i += stride) p[i] = v;
}

__global__ void k_relu(float* __restrict__ p, size_t n) {
    size_t i = (size_t)blockIdx.x * blockDim.x + threadIdx.x;
    size_t stride = (size_t)gridDim.x * blockDim.x;
    for (; i < n; i += stride) p[i] = fmaxf(p[i], 0.f);
}

__global__ void k_bias_init(float* __restrict__ x, const float* __restrict__ bias) {
    size_t n = (size_t)MTOT_ * ED_;
    size_t i = (size_t)blockIdx.x * blockDim.x + threadIdx.x;
    size_t stride = (size_t)gridDim.x * blockDim.x;
    for (; i < n; i += stride) x[i] = bias[i & (ED_ - 1)];
}

// ---------------------------------------------------------------- encoder
__global__ __launch_bounds__(256)
void k_encoder(const float* __restrict__ agvs, const float* __restrict__ stat,
               const float* __restrict__ bits,
               const float* __restrict__ We1, const float* __restrict__ be1,
               const float* __restrict__ We2, const float* __restrict__ be2,
               float* __restrict__ emb)
{
    __shared__ float srow[F_ + 2];
    __shared__ float sh[2 * ED_];
    const int row = blockIdx.x;             // 0 .. B*70-1
    const int b = row / ROWS_, i = row % ROWS_;
    const int t = threadIdx.x;
    if (t < F_)
        srow[t] = (i < A_) ? agvs[((size_t)b * A_ + i) * F_ + t]
                           : stat[((size_t)b * S_ + (i - A_)) * F_ + t];
    if (t < 2) srow[F_ + t] = bits[i * 2 + t];
    __syncthreads();
    {
        float acc = be1[t];
        #pragma unroll
        for (int k = 0; k < F_ + 2; ++k) acc += srow[k] * We1[k * (2 * ED_) + t];
        sh[t] = fmaxf(acc, 0.f);
    }
    __syncthreads();
    if (t < ED_) {
        float acc = be2[t];
        for (int k = 0; k < 2 * ED_; ++k) acc += sh[k] * We2[k * ED_ + t];
        emb[(size_t)row * ED_ + t] = fmaxf(acc, 0.f);
    }
}

__global__ void k_scatter_emb(const float* __restrict__ emb,
                              const int* __restrict__ node_idx,
                              float* __restrict__ x)
{
    const int row = blockIdx.x;            // B*70 blocks, 128 threads
    const int c = threadIdx.x;
    const int b = row / ROWS_;
    const int n = node_idx[row];
    if (n >= 0 && n < N_)
        atomAddF(&x[(((size_t)b * N_ + n) * ED_) + c], emb[(size_t)row * ED_ + c]);
}

__global__ void k_target(const int* __restrict__ tgt, float* __restrict__ x) {
    const int b = blockIdx.x, c = threadIdx.x;
    const int tn = tgt[b];
    if (tn >= 0 && tn < N_) x[(((size_t)b * N_ + tn) * ED_) + c] = 1.0f;
}

// ---------------------------------------------------------------- graph norm
__global__ void k_deg(const int* __restrict__ edge, float* __restrict__ deg) {
    const int e = blockIdx.x * blockDim.x + threadIdx.x;
    if (e >= NE_) return;
    const int d = (e < EDG_) ? edge[EDG_ + e] : (e - EDG_);
    atomAddF(&deg[d], 1.0f);
}

__global__ void k_norm(const int* __restrict__ edge, const float* __restrict__ deg,
                       float* __restrict__ nrm) {
    const int e = blockIdx.x * blockDim.x + threadIdx.x;
    if (e >= NE_) return;
    const int s = (e < EDG_) ? edge[e]        : (e - EDG_);
    const int d = (e < EDG_) ? edge[EDG_ + e] : (e - EDG_);
    nrm[e] = rsqrtf(deg[s]) * rsqrtf(deg[d]);
}

// ---------------------------------------------------------------- WMMA GEMM
// OUT[M,128] = X[M,128] @ W[128,128]; 4 waves/WG, 64-row tile, bf16 WMMA,
// A staged f32 via global_load_async_to_lds_b128 (double buffered, ASYNCcnt),
// W staged once, transposed+bf16, fragment reads are contiguous ds_load_b128.
#define TILE_M 64
#define AST    36     // f32 row stride of A slice (32 + 4 pad -> 36-dword stride)
#define WST    136    // bf16 col stride of W^T (128 + 8 pad -> 68-dword stride)

__global__ __launch_bounds__(128)
void k_gemm_wmma(const float* __restrict__ X, const float* __restrict__ W,
                 float* __restrict__ OUT)
{
    __shared__ alignas(16) float  sA[2][TILE_M * AST];  // 2 x 9 KB
    __shared__ alignas(16) __bf16 sW[128 * WST];        // 34 KB

    const int t = threadIdx.x;
    const int rowBase = blockIdx.x * TILE_M;

    // ---- stage W^T as bf16: sW[col*WST + k], coalesced global reads
    #pragma unroll
    for (int j = 0; j < 32; ++j) {
        const int i   = t + 128 * j;       // 0..4095 chunks of 4 k-values
        const int col = i & 127;
        const int k0  = (i >> 7) * 4;
        __bf16 t4[4];
        #pragma unroll
        for (int d = 0; d < 4; ++d) t4[d] = (__bf16)W[(size_t)(k0 + d) * ED_ + col];
        *(uint2*)&sW[col * WST + k0] = *(const uint2*)t4;
    }

    // ---- async stage of one 64x32 f32 A slice (4 x b128 per thread)
    auto stageA = [&](int buf, int kk) {
        #pragma unroll
        for (int j = 0; j < 4; ++j) {
            const int c  = t + 128 * j;    // 0..511 chunks of 4 floats
            const int rr = c >> 3;         // row in tile
            const int q  = c & 7;          // 16B chunk within the 32-float row
            const float* g = X + (size_t)(rowBase + rr) * ED_ + kk + q * 4;
            async_g2l_b128(lds_off(&sA[buf][rr * AST + q * 4]), g);
        }
    };

    stageA(0, 0);

    const int wave = t >> 5;
    const int lane = t & 31;
    const int half = lane >> 4;
    const int r    = lane & 15;

    v8f acc[8];
    const v8f vzero = {0.f, 0.f, 0.f, 0.f, 0.f, 0.f, 0.f, 0.f};
    #pragma unroll
    for (int n = 0; n < 8; ++n) acc[n] = vzero;

    #pragma unroll
    for (int s = 0; s < 4; ++s) {
        const int kk = s * 32;
        if (s < 3) {
            stageA((s + 1) & 1, kk + 32);                       // prefetch next slice
            asm volatile("s_wait_asynccnt 0x4" ::: "memory");   // current slice done
        } else {
            asm volatile("s_wait_asynccnt 0x0" ::: "memory");
        }
        __syncthreads();

        // A fragment: two contiguous 8-float runs per lane -> 4x ds_load_b128
        const float* arow = &sA[s & 1][(wave * 16 + r) * AST];
        const v4f a0 = *(const v4f*)(arow + half * 8);
        const v4f a1 = *(const v4f*)(arow + half * 8 + 4);
        const v4f a2 = *(const v4f*)(arow + 16 + half * 8);
        const v4f a3 = *(const v4f*)(arow + 16 + half * 8 + 4);
        v16bf afrag;
        #pragma unroll
        for (int q = 0; q < 4; ++q) {
            afrag[q]      = (__bf16)a0[q];
            afrag[4 + q]  = (__bf16)a1[q];
            afrag[8 + q]  = (__bf16)a2[q];
            afrag[12 + q] = (__bf16)a3[q];
        }

        // B fragments: two contiguous 8-bf16 runs per lane -> 2x ds_load_b128
        #pragma unroll
        for (int n = 0; n < 8; ++n) {
            const __bf16* wcol = &sW[(n * 16 + r) * WST + kk + half * 8];
            const v8bf b0 = *(const v8bf*)(wcol);
            const v8bf b1 = *(const v8bf*)(wcol + 16);
            const v16bf bfrag = __builtin_shufflevector(
                b0, b1, 0, 1, 2, 3, 4, 5, 6, 7, 8, 9, 10, 11, 12, 13, 14, 15);
            acc[n] = __builtin_amdgcn_wmma_f32_16x16x32_bf16(
                false, afrag, false, bfrag, (short)0, acc[n], false, false);
        }
        __syncthreads();   // all reads of this buffer done before it is re-staged
    }

    // C/D layout: VGPR g, lanes 0-15 -> (M=g, N=lane); lanes 16-31 -> (M=8+g)
    #pragma unroll
    for (int n = 0; n < 8; ++n) {
        #pragma unroll
        for (int g = 0; g < 8; ++g) {
            const int grow = rowBase + wave * 16 + g + half * 8;
            OUT[(size_t)grow * ED_ + n * 16 + r] = acc[n][g];
        }
    }
}

// ---------------------------------------------------------------- aggregation
__global__ void k_edge_scatter(const int* __restrict__ edge,
                               const float* __restrict__ nrm,
                               const float* __restrict__ lin,
                               float* __restrict__ x)
{
    const size_t total = (size_t)NE_ * B_ * (ED_ / 4);
    size_t idx = (size_t)blockIdx.x * blockDim.x + threadIdx.x;
    const size_t stride = (size_t)gridDim.x * blockDim.x;
    for (; idx < total; idx += stride) {
        const int c4 = (int)(idx & 31);
        const int b  = (int)((idx >> 5) & 63);
        const int e  = (int)(idx >> 11);
        const int s = (e < EDG_) ? edge[e]        : (e - EDG_);
        const int d = (e < EDG_) ? edge[EDG_ + e] : (e - EDG_);
        const float nm = nrm[e];
        const float4 v = *(const float4*)&lin[(((size_t)b * N_ + s) * ED_) + c4 * 4];
        float* dp = &x[(((size_t)b * N_ + d) * ED_) + c4 * 4];
        atomAddF(dp + 0, v.x * nm);
        atomAddF(dp + 1, v.y * nm);
        atomAddF(dp + 2, v.z * nm);
        atomAddF(dp + 3, v.w * nm);
    }
}

// ---------------------------------------------------------------- head MLP
__global__ __launch_bounds__(256)
void k_head(const float* __restrict__ x, const int* __restrict__ reach,
            const float* __restrict__ Wa1, const float* __restrict__ ba1,
            const float* __restrict__ Wa2, const float* __restrict__ ba2,
            const float* __restrict__ Wa3, const float* __restrict__ ba3,
            const float* __restrict__ mask, float* __restrict__ out)
{
    __shared__ float f[4 * ED_];
    __shared__ float a1[4 * ED_];
    __shared__ float a2[ED_];
    __shared__ float act[NOUT_];
    const int b = blockIdx.x;
    const int t = threadIdx.x;
    for (int j = t; j < 4 * ED_; j += 256) {
        const int slot = j >> 7, c = j & 127;
        const int node = reach[b * 4 + slot];
        f[j] = x[(((size_t)b * N_ + node) * ED_) + c];
    }
    __syncthreads();
    for (int j = t; j < 4 * ED_; j += 256) {
        float acc = ba1[j];
        for (int k = 0; k < 4 * ED_; ++k) acc += f[k] * Wa1[k * (4 * ED_) + j];
        a1[j] = fmaxf(acc, 0.f);
    }
    __syncthreads();
    if (t < ED_) {
        float acc = ba2[t];
        for (int k = 0; k < 4 * ED_; ++k) acc += a1[k] * Wa2[k * ED_ + t];
        a2[t] = fmaxf(acc, 0.f);
    }
    __syncthreads();
    if (t < NOUT_) {
        float acc = ba3[t];
        for (int k = 0; k < ED_; ++k) acc += a2[k] * Wa3[k * NOUT_ + t];
        act[t] = acc;
    }
    __syncthreads();
    for (int j = t; j < A_ * NOUT_; j += 256) {
        const int a = j >> 4, o = j & 15;
        const float v = (a == 0) ? act[o] : 0.f;
        out[(size_t)b * (A_ * NOUT_) + j] =
            v + (mask[(((size_t)b * A_ + a) * NOUT_) + o] - 1.0f) * 1.0e8f;
    }
}

// ---------------------------------------------------------------- launch
extern "C" void kernel_launch(void* const* d_in, const int* in_sizes, int n_in,
                              void* d_out, int out_size, void* d_ws, size_t ws_size,
                              hipStream_t stream) {
    (void)in_sizes; (void)n_in; (void)out_size; (void)ws_size;
    const float* agvs = (const float*)d_in[0];
    const float* stat = (const float*)d_in[1];
    const float* mask = (const float*)d_in[2];
    const int*   nidx = (const int*)d_in[3];
    const int*   tgt  = (const int*)d_in[4];
    const int*   rch  = (const int*)d_in[5];
    const int*   edge = (const int*)d_in[6];
    const float* bits = (const float*)d_in[7];
    const float* We1  = (const float*)d_in[8];
    const float* be1  = (const float*)d_in[9];
    const float* We2  = (const float*)d_in[10];
    const float* be2  = (const float*)d_in[11];
    const float* gW   = (const float*)d_in[12];
    const float* gb   = (const float*)d_in[13];
    const float* Wa1  = (const float*)d_in[14];
    const float* ba1  = (const float*)d_in[15];
    const float* Wa2  = (const float*)d_in[16];
    const float* ba2  = (const float*)d_in[17];
    const float* Wa3  = (const float*)d_in[18];
    const float* ba3  = (const float*)d_in[19];
    float* out = (float*)d_out;

    float* ws  = (float*)d_ws;
    float* x   = ws;                                  // MTOT_*ED_  (118 MB)
    float* tmp = x   + (size_t)MTOT_ * ED_;           // MTOT_*ED_  (118 MB)
    float* emb = tmp + (size_t)MTOT_ * ED_;           // B*70*128
    float* deg = emb + (size_t)B_ * ROWS_ * ED_;      // N
    float* nrm = deg + N_;                            // NE_

    const size_t xelems = (size_t)MTOT_ * ED_;

    k_fill<<<2048, 256, 0, stream>>>(x, xelems, 0.f);
    k_fill<<<16, 256, 0, stream>>>(deg, (size_t)N_, 0.f);

    k_encoder<<<B_ * ROWS_, 256, 0, stream>>>(agvs, stat, bits, We1, be1, We2, be2, emb);
    k_scatter_emb<<<B_ * ROWS_, ED_, 0, stream>>>(emb, nidx, x);
    k_target<<<B_, ED_, 0, stream>>>(tgt, x);

    k_deg<<<(NE_ + 255) / 256, 256, 0, stream>>>(edge, deg);
    k_norm<<<(NE_ + 255) / 256, 256, 0, stream>>>(edge, deg, nrm);

    for (int l = 0; l < 6; ++l) {
        k_gemm_wmma<<<MTOT_ / TILE_M, 128, 0, stream>>>(x, gW + (size_t)l * ED_ * ED_, tmp);
        k_bias_init<<<2048, 256, 0, stream>>>(x, gb + (size_t)l * ED_);
        k_edge_scatter<<<4096, 256, 0, stream>>>(edge, nrm, tmp, x);
        k_relu<<<2048, 256, 0, stream>>>(x, xelems);
    }

    k_head<<<B_, 256, 0, stream>>>(x, rch, Wa1, ba1, Wa2, ba2, Wa3, ba3, mask, out);
}
